// GAT_80290118632018
// MI455X (gfx1250) — matrix operand order repaired and checked by reading, not compiled, not run on hardware
//
#include <hip/hip_runtime.h>
#include <hip/hip_bf16.h>

#define NNODE 50000
#define NEDGE 800000
#define FIN   256
#define FHID  64
#define FOUT  64
#define NH1   4
#define NH2   1
#define NEG_SLOPE 0.2f

typedef __bf16 v16bf __attribute__((ext_vector_type(16)));
typedef __bf16 v8bf  __attribute__((ext_vector_type(8)));
typedef float  v8f   __attribute__((ext_vector_type(8)));

// ---------------------------------------------------------------- utilities
__global__ void fill_f32(float* __restrict__ p, float v, int n) {
    int i = blockIdx.x * blockDim.x + threadIdx.x;
    if (i < n) p[i] = v;
}

__global__ void f32_to_bf16(const float* __restrict__ in, __bf16* __restrict__ out, int n) {
    int i = blockIdx.x * blockDim.x + threadIdx.x;
    if (i < n) out[i] = (__bf16)in[i];
}

__device__ __forceinline__ void atomicMaxF(float* addr, float val) {
    int* ai = (int*)addr;
    int old = __float_as_int(*addr);
    while (__int_as_float(old) < val) {
        int assumed = old;
        old = atomicCAS(ai, assumed, __float_as_int(val));
        if (old == assumed) break;
    }
}

// Convert W (K x Nout, row-major f32) -> Bt (Nout x K, row-major bf16) so the
// WMMA B-fragment is 16 contiguous bf16 (one aligned 32B load) per lane.
__global__ void convert_wT(const float* __restrict__ W, __bf16* __restrict__ Bt,
                           int K, int Nout) {
    int i = blockIdx.x * blockDim.x + threadIdx.x;
    if (i >= K * Nout) return;
    int k = i / Nout;
    int n = i - k * Nout;
    Bt[(size_t)n * K + k] = (__bf16)W[i];
}

// ---------------------------------------------------------------- WMMA GEMM
// C[M x Nout] = A[M x K, bf16] * Bt^T (Bt is Nout x K bf16), f32 accumulate.
// One wave per 16x16 tile; inner loop: 2 b128 A-loads + 1 b256 B-load + WMMA.
__global__ void gemm_wmma_bf16(const __bf16* __restrict__ A,
                               const __bf16* __restrict__ Bt,
                               float* __restrict__ C,
                               int M, int K, int Nout) {
    const int lane   = threadIdx.x & 31;
    const int wave   = (blockIdx.x * blockDim.x + threadIdx.x) >> 5;
    const int tilesN = Nout >> 4;
    const int tm     = wave / tilesN;
    const int tn     = wave - tm * tilesN;
    if (tm * 16 >= M) return;

    const int l16 = lane & 15;
    const int hi  = lane >> 4;              // 0: lanes 0-15, 1: lanes 16-31
    const int kA  = hi * 16;                // A lane K sub-offset (contiguous k = 16*hi + s)
    const int kB  = hi * 16;                // B lane K sub-offset (contiguous k = 16*hi + s)

    const __bf16* ap = A  + (size_t)(tm * 16 + l16) * K + kA;
    const __bf16* bp = Bt + (size_t)(tn * 16 + l16) * K + kB;

    v8f acc = {};
    for (int k0 = 0; k0 < K; k0 += 32) {
        v8bf alo = *(const v8bf*)(ap + k0);        // K = kA .. kA+7
        v8bf ahi = *(const v8bf*)(ap + k0 + 8);    // K = kA+8 .. kA+15
        v16bf af = __builtin_shufflevector(alo, ahi,
                     0, 1, 2, 3, 4, 5, 6, 7, 8, 9, 10, 11, 12, 13, 14, 15);
        v16bf bf = *(const v16bf*)(bp + k0);
        acc = __builtin_amdgcn_wmma_f32_16x16x32_bf16(
            /*neg_a=*/false, af, /*neg_b=*/false, bf,
            /*c_mod=*/(short)0, acc, /*reuse_a=*/false, /*reuse_b=*/false);
    }
    // C/D layout: VGPR v holds row (v + 8*hi), column l16
    float* cp = C + (size_t)(tm * 16 + hi * 8) * Nout + tn * 16 + l16;
    #pragma unroll
    for (int v = 0; v < 8; ++v) cp[(size_t)v * Nout] = acc[v];
}

// ---------------------------------------------------------------- attention
// el[n,h] = sum_d feat[n,h,d]*al[h,d];  er likewise. One wave per (node,head).
__global__ void attn_scores(const float* __restrict__ feat,
                            const float* __restrict__ al,
                            const float* __restrict__ ar,
                            float* __restrict__ el, float* __restrict__ er,
                            int Nn, int H, int D) {
    const int lane = threadIdx.x & 31;
    const int wid  = (blockIdx.x * blockDim.x + threadIdx.x) >> 5;
    const int n = wid / H, h = wid - n * H;
    if (n >= Nn) return;
    const float* f = feat + (size_t)n * H * D + h * D;
    float sl = 0.f, sr = 0.f;
    for (int d = lane; d < D; d += 32) {
        float v = f[d];
        sl += v * al[h * D + d];
        sr += v * ar[h * D + d];
    }
    #pragma unroll
    for (int off = 16; off > 0; off >>= 1) {
        sl += __shfl_xor(sl, off, 32);
        sr += __shfl_xor(sr, off, 32);
    }
    if (lane == 0) { el[n * H + h] = sl; er[n * H + h] = sr; }
}

// e = leaky_relu(el[src] + er[dst]); segment-max by dst via float atomic max.
__global__ void edge_logits(const int* __restrict__ src, const int* __restrict__ dst,
                            const float* __restrict__ el, const float* __restrict__ er,
                            float* __restrict__ eout, float* __restrict__ emax,
                            int Etot, int H) {
    int i = blockIdx.x * blockDim.x + threadIdx.x;
    if (i >= Etot * H) return;
    int e = i / H, h = i - e * H;
    int s = src[e], d = dst[e];
    float v = el[s * H + h] + er[d * H + h];
    v = (v >= 0.f) ? v : NEG_SLOPE * v;
    eout[i] = v;
    atomicMaxF(&emax[d * H + h], v);
}

// ee = exp(e - emax[dst]); segment-sum by dst.
__global__ void edge_exp(const int* __restrict__ dst,
                         float* __restrict__ eout, const float* __restrict__ emax,
                         float* __restrict__ esum, int Etot, int H) {
    int i = blockIdx.x * blockDim.x + threadIdx.x;
    if (i >= Etot * H) return;
    int e = i / H, h = i - e * H;
    int d = dst[e];
    float ex = expf(eout[i] - emax[d * H + h]);
    eout[i] = ex;
    atomicAdd(&esum[d * H + h], ex);
}

// out[dst] += (ee/esum[dst]) * feat[src].  One block per edge, thread = (h,d).
__global__ void edge_aggregate(const int* __restrict__ src, const int* __restrict__ dst,
                               const float* __restrict__ feat,
                               const float* __restrict__ ee, const float* __restrict__ esum,
                               float* __restrict__ out, int H, int D) {
    int e = blockIdx.x;
    int t = threadIdx.x;                    // blockDim.x == H*D
    int h = t / D;
    int s = src[e], d = dst[e];
    float alpha = ee[(size_t)e * H + h] / esum[(size_t)d * H + h];
    atomicAdd(&out[(size_t)d * H * D + t], alpha * feat[(size_t)s * H * D + t]);
}

// (agg + bias) -> relu -> bf16 (fused down-convert producing layer-2 GEMM input)
__global__ void bias_relu_to_bf16(const float* __restrict__ in, const float* __restrict__ b,
                                  __bf16* __restrict__ outbf, int total, int HD) {
    int i = blockIdx.x * blockDim.x + threadIdx.x;
    if (i >= total) return;
    float v = in[i] + b[i % HD];
    outbf[i] = (__bf16)fmaxf(v, 0.f);
}

__global__ void bias_act(float* __restrict__ io, const float* __restrict__ b,
                         int total, int HD, int do_relu) {
    int i = blockIdx.x * blockDim.x + threadIdx.x;
    if (i >= total) return;
    float v = io[i] + b[i % HD];
    io[i] = do_relu ? fmaxf(v, 0.f) : v;
}

// ---------------------------------------------------------------- launcher
extern "C" void kernel_launch(void* const* d_in, const int* in_sizes, int n_in,
                              void* d_out, int out_size, void* d_ws, size_t ws_size,
                              hipStream_t stream) {
    (void)in_sizes; (void)n_in; (void)out_size; (void)ws_size;
    const float* features = (const float*)d_in[0];
    const int*   src      = (const int*)d_in[1];
    const int*   dst      = (const int*)d_in[2];
    const float* W1       = (const float*)d_in[3];
    const float* al1      = (const float*)d_in[4];
    const float* ar1      = (const float*)d_in[5];
    const float* b1       = (const float*)d_in[6];
    const float* W2       = (const float*)d_in[7];
    const float* al2      = (const float*)d_in[8];
    const float* ar2      = (const float*)d_in[9];
    const float* b2       = (const float*)d_in[10];
    float* out = (float*)d_out;

    char* ws = (char*)d_ws;
    size_t off = 0;
    auto alloc = [&](size_t bytes) -> char* {
        char* p = ws + off;
        off = (off + bytes + 255) & ~(size_t)255;
        return p;
    };
    float*  feat1 = (float*) alloc((size_t)NNODE * NH1 * FHID * 4);   // 51.2 MB
    float*  h1    = (float*) alloc((size_t)NNODE * NH1 * FHID * 4);   // 51.2 MB
    float*  feat2 = (float*) alloc((size_t)NNODE * FOUT * 4);         // 12.8 MB
    __bf16* Abf   = (__bf16*)alloc((size_t)NNODE * FIN * 2);          // 25.6 MB (shared: X_bf16 then h1_bf16)
    __bf16* W1t   = (__bf16*)alloc((size_t)FIN * NH1 * FHID * 2);
    __bf16* W2t   = (__bf16*)alloc((size_t)NH1 * FHID * NH2 * FOUT * 2);
    float*  el1   = (float*) alloc((size_t)NNODE * NH1 * 4);
    float*  er1   = (float*) alloc((size_t)NNODE * NH1 * 4);
    float*  emax1 = (float*) alloc((size_t)NNODE * NH1 * 4);
    float*  esum1 = (float*) alloc((size_t)NNODE * NH1 * 4);
    float*  ee1   = (float*) alloc((size_t)NEDGE * NH1 * 4);          // 12.8 MB
    float*  el2   = (float*) alloc((size_t)NNODE * 4);
    float*  er2   = (float*) alloc((size_t)NNODE * 4);
    float*  emax2 = (float*) alloc((size_t)NNODE * 4);
    float*  esum2 = (float*) alloc((size_t)NNODE * 4);
    float*  ee2   = (float*) alloc((size_t)NEDGE * 4);                // 3.2 MB

    const float NEGINF = -__builtin_huge_valf();
    auto cdiv = [](int a, int b) { return (a + b - 1) / b; };

    // -- weight + activation prep (bf16) ----------------------------------
    convert_wT<<<cdiv(FIN * NH1 * FHID, 256), 256, 0, stream>>>(W1, W1t, FIN, NH1 * FHID);
    convert_wT<<<cdiv(NH1 * FHID * NH2 * FOUT, 256), 256, 0, stream>>>(W2, W2t, NH1 * FHID, NH2 * FOUT);
    f32_to_bf16<<<cdiv(NNODE * FIN, 256), 256, 0, stream>>>(features, Abf, NNODE * FIN);

    // ===================== Layer 1 (H=4, D=64, relu) =====================
    {
        const int M = NNODE, K = FIN, Nout = NH1 * FHID;   // 50000 x 256 x 256
        int waves = (M / 16) * (Nout / 16);                // 50000 waves
        gemm_wmma_bf16<<<cdiv(waves * 32, 128), 128, 0, stream>>>(Abf, W1t, feat1, M, K, Nout);

        int aw = NNODE * NH1;                              // one wave per (node,head)
        attn_scores<<<cdiv(aw * 32, 256), 256, 0, stream>>>(feat1, al1, ar1, el1, er1, NNODE, NH1, FHID);

        fill_f32<<<cdiv(NNODE * NH1, 256), 256, 0, stream>>>(emax1, NEGINF, NNODE * NH1);
        fill_f32<<<cdiv(NNODE * NH1, 256), 256, 0, stream>>>(esum1, 0.f, NNODE * NH1);
        fill_f32<<<cdiv(NNODE * NH1 * FHID, 256), 256, 0, stream>>>(h1, 0.f, NNODE * NH1 * FHID);

        edge_logits<<<cdiv(NEDGE * NH1, 256), 256, 0, stream>>>(src, dst, el1, er1, ee1, emax1, NEDGE, NH1);
        edge_exp<<<cdiv(NEDGE * NH1, 256), 256, 0, stream>>>(dst, ee1, emax1, esum1, NEDGE, NH1);
        edge_aggregate<<<NEDGE, NH1 * FHID, 0, stream>>>(src, dst, feat1, ee1, esum1, h1, NH1, FHID);

        // fused: (h1 + b1) -> relu -> bf16 into Abf (layer-2 GEMM input)
        bias_relu_to_bf16<<<cdiv(NNODE * NH1 * FHID, 256), 256, 0, stream>>>(
            h1, b1, Abf, NNODE * NH1 * FHID, NH1 * FHID);
    }

    // ===================== Layer 2 (H=1, D=64, no relu) ==================
    {
        const int M = NNODE, K = NH1 * FHID, Nout = NH2 * FOUT;  // 50000 x 256 x 64
        int waves = (M / 16) * (Nout / 16);                       // 12500 waves
        gemm_wmma_bf16<<<cdiv(waves * 32, 128), 128, 0, stream>>>(Abf, W2t, feat2, M, K, Nout);

        int aw = NNODE * NH2;
        attn_scores<<<cdiv(aw * 32, 256), 256, 0, stream>>>(feat2, al2, ar2, el2, er2, NNODE, NH2, FOUT);

        fill_f32<<<cdiv(NNODE, 256), 256, 0, stream>>>(emax2, NEGINF, NNODE);
        fill_f32<<<cdiv(NNODE, 256), 256, 0, stream>>>(esum2, 0.f, NNODE);
        fill_f32<<<cdiv(NNODE * FOUT, 256), 256, 0, stream>>>(out, 0.f, NNODE * FOUT);

        edge_logits<<<cdiv(NEDGE, 256), 256, 0, stream>>>(src, dst, el2, er2, ee2, emax2, NEDGE, NH2);
        edge_exp<<<cdiv(NEDGE, 256), 256, 0, stream>>>(dst, ee2, emax2, esum2, NEDGE, NH2);
        edge_aggregate<<<NEDGE, NH2 * FOUT, 0, stream>>>(src, dst, feat2, ee2, esum2, out, NH2, FOUT);

        // mean over H2=1 heads is identity; add (zero) bias, no relu
        bias_act<<<cdiv(NNODE * FOUT, 256), 256, 0, stream>>>(out, b2, NNODE * FOUT, NH2 * FOUT, 0);
    }
}
